// SpiralAutoencoder_14645838479703
// MI455X (gfx1250) — compile-verified, matrix operand order checked by reading, not run on hardware
//
#include <hip/hip_runtime.h>
#include <math.h>

typedef __attribute__((ext_vector_type(16))) _Float16 v16h;
typedef __attribute__((ext_vector_type(8)))  float    v8f;
typedef __attribute__((ext_vector_type(4)))  float    f32x4;

#define WAVES_PER_BLOCK 8

static __device__ __forceinline__ int imin_(int a, int b) { return a < b ? a : b; }
static __device__ __forceinline__ float sigmoidf_(float x) { return 1.0f / (1.0f + expf(-x)); }
static __device__ __forceinline__ float eluf_(float x) { return x > 0.0f ? x : (expf(x) - 1.0f); }

// Fragment fill, row-major contiguous: lane (grp,mn) holds K-segments
// [k0+8*grp, +8) and [k0+16+8*grp, +8) of its row.  4x b128 loads, no guards.
static __device__ __forceinline__ v16h frag_row(const float* __restrict__ row,
                                                int k0, int grp) {
  v16h f;
  #pragma unroll
  for (int j = 0; j < 2; ++j) {
    const int kseg = k0 + j * 16 + grp * 8;
    const f32x4 p0 = *(const f32x4*)(row + kseg);
    const f32x4 p1 = *(const f32x4*)(row + kseg + 4);
    #pragma unroll
    for (int e = 0; e < 4; ++e) {
      f[j * 8 + e]     = (_Float16)p0[e];
      f[j * 8 + 4 + e] = (_Float16)p1[e];
    }
  }
  return f;
}

// Fragment fill, spiral gather: requires Cin % 8 == 0 so each 8-wide segment
// stays within one spiral neighbor.
static __device__ __forceinline__ v16h frag_gather(const float* __restrict__ xBase,
                                                   const int* __restrict__ idxRow,
                                                   int Cin, int k0, int grp) {
  v16h f;
  #pragma unroll
  for (int j = 0; j < 2; ++j) {
    const int kseg = k0 + j * 16 + grp * 8;
    const int s = kseg / Cin;
    const int ci = kseg - s * Cin;
    const float* src = xBase + (size_t)idxRow[s] * Cin + ci;
    const f32x4 p0 = *(const f32x4*)(src);
    const f32x4 p1 = *(const f32x4*)(src + 4);
    #pragma unroll
    for (int e = 0; e < 4; ++e) {
      f[j * 8 + e]     = (_Float16)p0[e];
      f[j * 8 + 4 + e] = (_Float16)p1[e];
    }
  }
  return f;
}

#define WMMA_F16(a, b, c) \
  __builtin_amdgcn_wmma_f32_16x16x32_f16(false, (a), false, (b), (short)0, (c), false, false)

// ---------------------------------------------------------------------------
// WMMA GEMM: out[M,N] = A[M,K] @ W[N,K]^T + bias[N].  K % 32 == 0 required.
// Each wave computes a 2x2 block of 16x16 tiles: 4 independent accumulator
// chains per K-step off 4 fragment loads (16 b32 loads per v_wmma).
// ---------------------------------------------------------------------------
__global__ __launch_bounds__(256)
void wmma_gemm_nt(const float* __restrict__ A, const float* __restrict__ W,
                  const float* __restrict__ bias, float* __restrict__ out,
                  int M, int N, int K) {
  const int wave = threadIdx.x >> 5;
  const int lane = threadIdx.x & 31;
  const int grp = lane >> 4;
  const int mn = lane & 15;

  const int nTileN = (N + 15) >> 4;
  const int nTileM = (M + 15) >> 4;
  const int nPairN = (nTileN + 1) >> 1;
  const int nPairM = (nTileM + 1) >> 1;
  const int totalPairs = nPairM * nPairN;
  int pair = blockIdx.x * WAVES_PER_BLOCK + wave;
  const bool active = pair < totalPairs;
  if (!active) pair = totalPairs - 1;
  const int tm0 = (pair / nPairN) * 2;
  const int tn0 = (pair % nPairN) * 2;
  const int tm1 = imin_(tm0 + 1, nTileM - 1);
  const int tn1 = imin_(tn0 + 1, nTileN - 1);

  // Clamped row pointers: clamped (duplicate) rows only feed unstored tiles.
  const float* aRow0 = A + (size_t)imin_(tm0 * 16 + mn, M - 1) * K;
  const float* aRow1 = A + (size_t)imin_(tm1 * 16 + mn, M - 1) * K;
  const float* wRow0 = W + (size_t)imin_(tn0 * 16 + mn, N - 1) * K;
  const float* wRow1 = W + (size_t)imin_(tn1 * 16 + mn, N - 1) * K;

  v8f acc00 = {}, acc01 = {}, acc10 = {}, acc11 = {};
  const int ksteps = K >> 5;
  for (int ks = 0; ks < ksteps; ++ks) {
    const int k0 = ks * 32;
    __builtin_prefetch(wRow0 + k0 + 128, 0, 1);
    const v16h a0 = frag_row(aRow0, k0, grp);
    const v16h a1 = frag_row(aRow1, k0, grp);
    const v16h b0 = frag_row(wRow0, k0, grp);
    const v16h b1 = frag_row(wRow1, k0, grp);
    acc00 = WMMA_F16(a0, b0, acc00);
    acc01 = WMMA_F16(a0, b1, acc01);
    acc10 = WMMA_F16(a1, b0, acc10);
    acc11 = WMMA_F16(a1, b1, acc11);
  }

  const bool sM1 = tm1 > tm0;
  const bool sN1 = tn1 > tn0;
  const int co0 = tn0 * 16 + mn;
  const int co1 = tn1 * 16 + mn;
  #pragma unroll
  for (int r = 0; r < 8; ++r) {
    const int m0 = tm0 * 16 + r + grp * 8;
    const int m1 = tm1 * 16 + r + grp * 8;
    if (active) {
      if (m0 < M && co0 < N)               out[(size_t)m0 * N + co0] = acc00[r] + bias[co0];
      if (sN1 && m0 < M && co1 < N)        out[(size_t)m0 * N + co1] = acc01[r] + bias[co1];
      if (sM1 && m1 < M && co0 < N)        out[(size_t)m1 * N + co0] = acc10[r] + bias[co0];
      if (sM1 && sN1 && m1 < M && co1 < N) out[(size_t)m1 * N + co1] = acc11[r] + bias[co1];
    }
  }
}

// ---------------------------------------------------------------------------
// WMMA spiral conv (fast): Cin % 8 == 0 and K = 9*Cin % 32 == 0 required.
// out[b,v,co] = act( sum_{s,ci} x[b, idx[v,s], ci] * W[co, s*Cin+ci]
//                    + bias[co] ) (+ residual[v,co])
// Same 2x2 tile blocking as the GEMM.
// ---------------------------------------------------------------------------
__global__ __launch_bounds__(256)
void wmma_spiral_conv(const float* __restrict__ x, const int* __restrict__ idx,
                      const float* __restrict__ W, const float* __restrict__ bias,
                      const float* __restrict__ residual, float* __restrict__ out,
                      int Bf, int Nv, int Cin, int Cout, int applyElu) {
  const int wave = threadIdx.x >> 5;
  const int lane = threadIdx.x & 31;
  const int grp = lane >> 4;
  const int mn = lane & 15;

  const int M = Bf * Nv;
  const int K = 9 * Cin;
  const int nTileN = (Cout + 15) >> 4;
  const int nTileM = (M + 15) >> 4;
  const int nPairN = (nTileN + 1) >> 1;
  const int nPairM = (nTileM + 1) >> 1;
  const int totalPairs = nPairM * nPairN;
  int pair = blockIdx.x * WAVES_PER_BLOCK + wave;
  const bool active = pair < totalPairs;
  if (!active) pair = totalPairs - 1;
  const int tm0 = (pair / nPairN) * 2;
  const int tn0 = (pair % nPairN) * 2;
  const int tm1 = imin_(tm0 + 1, nTileM - 1);
  const int tn1 = imin_(tn0 + 1, nTileN - 1);

  const int row0 = imin_(tm0 * 16 + mn, M - 1);
  const int row1 = imin_(tm1 * 16 + mn, M - 1);
  const int b0i = row0 / Nv, v0i = row0 - b0i * Nv;
  const int b1i = row1 / Nv, v1i = row1 - b1i * Nv;
  const int* __restrict__ idxRow0 = idx + (size_t)v0i * 9;
  const int* __restrict__ idxRow1 = idx + (size_t)v1i * 9;
  const float* __restrict__ xBase0 = x + (size_t)b0i * Nv * Cin;
  const float* __restrict__ xBase1 = x + (size_t)b1i * Nv * Cin;
  const float* __restrict__ wRow0 = W + (size_t)imin_(tn0 * 16 + mn, Cout - 1) * K;
  const float* __restrict__ wRow1 = W + (size_t)imin_(tn1 * 16 + mn, Cout - 1) * K;

  v8f acc00 = {}, acc01 = {}, acc10 = {}, acc11 = {};
  const int ksteps = K >> 5;
  for (int ks = 0; ks < ksteps; ++ks) {
    const int k0 = ks * 32;
    __builtin_prefetch(wRow0 + k0 + 128, 0, 1);
    const v16h a0 = frag_gather(xBase0, idxRow0, Cin, k0, grp);
    const v16h a1 = frag_gather(xBase1, idxRow1, Cin, k0, grp);
    const v16h b0 = frag_row(wRow0, k0, grp);
    const v16h b1 = frag_row(wRow1, k0, grp);
    acc00 = WMMA_F16(a0, b0, acc00);
    acc01 = WMMA_F16(a0, b1, acc01);
    acc10 = WMMA_F16(a1, b0, acc10);
    acc11 = WMMA_F16(a1, b1, acc11);
  }

  const bool sM1 = tm1 > tm0;
  const bool sN1 = tn1 > tn0;
  const int co0 = tn0 * 16 + mn;
  const int co1 = tn1 * 16 + mn;
  #pragma unroll
  for (int r = 0; r < 8; ++r) {
    const int m0 = tm0 * 16 + r + grp * 8;
    const int m1 = tm1 * 16 + r + grp * 8;
    #pragma unroll
    for (int q = 0; q < 4; ++q) {
      const int mrow = (q & 2) ? m1 : m0;
      const int co = (q & 1) ? co1 : co0;
      const bool en = active && ((q & 2) ? sM1 : true) && ((q & 1) ? sN1 : true) &&
                      mrow < M && co < Cout;
      if (en) {
        const v8f* pacc = (q == 0) ? &acc00 : (q == 1) ? &acc01 : (q == 2) ? &acc10 : &acc11;
        float v = (*pacc)[r] + bias[co];
        if (applyElu) v = eluf_(v);
        if (residual) {
          const int vb = mrow / Nv;
          const int vv = mrow - vb * Nv;
          v += residual[(size_t)vv * Cout + co];
        }
        out[(size_t)mrow * Cout + co] = v;
      }
    }
  }
}

// ---------------------------------------------------------------------------
// WMMA spiral conv (small/masked): any Cin, any K.  Used only for the
// encoder's first conv (Cin=3, K=27) -- negligible work.
// ---------------------------------------------------------------------------
__global__ __launch_bounds__(256)
void wmma_spiral_conv_small(const float* __restrict__ x, const int* __restrict__ idx,
                            const float* __restrict__ W, const float* __restrict__ bias,
                            float* __restrict__ out,
                            int Bf, int Nv, int Cin, int Cout, int applyElu) {
  const int wave = threadIdx.x >> 5;
  const int lane = threadIdx.x & 31;
  const int grp = lane >> 4;
  const int mn = lane & 15;

  const int M = Bf * Nv;
  const int K = 9 * Cin;
  const int nTileN = (Cout + 15) >> 4;
  const int nTileM = (M + 15) >> 4;
  const int totalTiles = nTileM * nTileN;
  int tile = blockIdx.x * WAVES_PER_BLOCK + wave;
  const bool active = tile < totalTiles;
  if (!active) tile = totalTiles - 1;
  const int tm = tile / nTileN;
  const int tn = tile % nTileN;

  const int rowC = imin_(tm * 16 + mn, M - 1);
  const int bIdx = rowC / Nv;
  const int vIdx = rowC - bIdx * Nv;
  const int* __restrict__ idxRow = idx + (size_t)vIdx * 9;
  const float* __restrict__ xBase = x + (size_t)bIdx * Nv * Cin;
  const float* __restrict__ wRow = W + (size_t)imin_(tn * 16 + mn, Cout - 1) * K;

  v8f acc = {};
  const int ksteps = (K + 31) >> 5;
  for (int ks = 0; ks < ksteps; ++ks) {
    const int k0 = ks * 32;
    v16h af, bf;
    #pragma unroll
    for (int h = 0; h < 16; ++h) {
      const int kk = ((h & 8) << 1) | (grp << 3) | (h & 7);
      const int k = k0 + kk;
      _Float16 av = (_Float16)0.0f, wv = (_Float16)0.0f;
      if (k < K) {
        const int s = k / Cin;
        const int ci = k - s * Cin;
        av = (_Float16)xBase[(size_t)idxRow[s] * Cin + ci];
        wv = (_Float16)wRow[k];
      }
      af[h] = av;
      bf[h] = wv;
    }
    acc = WMMA_F16(af, bf, acc);
  }

  const int co = tn * 16 + mn;
  #pragma unroll
  for (int r = 0; r < 8; ++r) {
    const int mrow = tm * 16 + r + grp * 8;
    if (active && mrow < M && co < Cout) {
      float v = acc[r] + bias[co];
      if (applyElu) v = eluf_(v);
      out[(size_t)mrow * Cout + co] = v;
    }
  }
}

// ---------------------------------------------------------------------------
// Sparse pool: out[b, rows[e], c] += in[b, cols[e], c] * vals[e]
// ---------------------------------------------------------------------------
__global__ __launch_bounds__(256)
void fill_zero(float* __restrict__ p, int n) {
  const int i = blockIdx.x * blockDim.x + threadIdx.x;
  if (i < n) p[i] = 0.0f;
}

__global__ __launch_bounds__(256)
void pool_scatter(const float* __restrict__ in, const int* __restrict__ rows,
                  const int* __restrict__ cols, const float* __restrict__ vals,
                  float* __restrict__ out, int nnz, int C, int Bf, int Nin, int Nout) {
  const int t = blockIdx.x * blockDim.x + threadIdx.x;
  if (t >= nnz * C) return;
  const int e = t / C;
  const int c = t - e * C;
  const int r = rows[e];
  const int col = cols[e];
  const float v = vals[e];
  for (int b = 0; b < Bf; ++b) {
    atomicAdd(&out[((size_t)b * Nout + r) * C + c],
              in[((size_t)b * Nin + col) * C + c] * v);
  }
}

// ---------------------------------------------------------------------------
// Concat: h_in[t, 0:64] = audio_emb[t]; h_in[t, 64:128] = z
// ---------------------------------------------------------------------------
__global__ __launch_bounds__(256)
void concat_kernel(const float* __restrict__ ae, const float* __restrict__ z,
                   float* __restrict__ h_in) {
  const int i = blockIdx.x * blockDim.x + threadIdx.x;
  if (i < 32 * 128) {
    const int t = i >> 7;
    const int c = i & 127;
    h_in[i] = (c < 64) ? ae[t * 64 + c] : z[c - 64];
  }
}

// ---------------------------------------------------------------------------
// 3-layer bidirectional LSTM, T=32, B=1, H=32, in one workgroup.
// Threads 0..127 = forward gates, 128..255 = reverse gates.
// ---------------------------------------------------------------------------
__global__ __launch_bounds__(256)
void lstm_stack(const float* __restrict__ h_in,
                const float* __restrict__ l0_Wih, const float* __restrict__ l0_Whh,
                const float* __restrict__ l0_bih, const float* __restrict__ l0_bhh,
                const float* __restrict__ l12_Wih, const float* __restrict__ l12_Whh,
                const float* __restrict__ l12_bih, const float* __restrict__ l12_bhh,
                float* __restrict__ latent) {
  __shared__ float xbuf[32][128];
  __shared__ float gbuf[256];
  __shared__ float hf[32], cf[32], hrv[32], crv[32];
  __shared__ float outf[32][32], outr[32][32];
  const int tid = threadIdx.x;

  for (int i = tid; i < 32 * 128; i += 256) xbuf[i >> 7][i & 127] = h_in[i];
  __syncthreads();

  for (int layer = 0; layer < 3; ++layer) {
    const int D = (layer == 0) ? 128 : 64;
    const int dir = tid >> 7;
    const int gi = tid & 127;
    const float *Wih, *Whh, *bih, *bhh;
    if (layer == 0) {
      Wih = l0_Wih + dir * 128 * 128;
      Whh = l0_Whh + dir * 128 * 32;
      bih = l0_bih + dir * 128;
      bhh = l0_bhh + dir * 128;
    } else {
      const int o = (layer - 1) * 2 + dir;
      Wih = l12_Wih + o * 128 * 64;
      Whh = l12_Whh + o * 128 * 32;
      bih = l12_bih + o * 128;
      bhh = l12_bhh + o * 128;
    }
    if (tid < 32) { hf[tid] = 0.0f; cf[tid] = 0.0f; hrv[tid] = 0.0f; crv[tid] = 0.0f; }
    __syncthreads();

    for (int t = 0; t < 32; ++t) {
      const int te = dir ? (31 - t) : t;
      float g = bih[gi] + bhh[gi];
      const float* wr = Wih + gi * D;
      for (int k = 0; k < D; ++k) g += wr[k] * xbuf[te][k];
      const float* hprev = dir ? hrv : hf;
      const float* wh = Whh + gi * 32;
      for (int k = 0; k < 32; ++k) g += wh[k] * hprev[k];
      gbuf[tid] = g;
      __syncthreads();
      if ((tid & 127) < 32) {
        const int j = tid & 31;
        const int d = tid >> 7;
        const int base = d * 128;
        const float iv = gbuf[base + j];
        const float fv = gbuf[base + 32 + j];
        const float gv = gbuf[base + 64 + j];
        const float ov = gbuf[base + 96 + j];
        const float cp = d ? crv[j] : cf[j];
        const float c = sigmoidf_(fv) * cp + sigmoidf_(iv) * tanhf(gv);
        const float h = sigmoidf_(ov) * tanhf(c);
        if (d) { crv[j] = c; hrv[j] = h; outr[31 - t][j] = h; }
        else   { cf[j]  = c; hf[j]  = h; outf[t][j] = h; }
      }
      __syncthreads();
    }

    for (int i = tid; i < 32 * 64; i += 256) {
      const int t = i >> 6;
      const int c = i & 63;
      xbuf[t][c] = (c < 32) ? outf[t][c] : outr[t][c - 32];
    }
    __syncthreads();
  }

  for (int i = tid; i < 32 * 64; i += 256) latent[i] = xbuf[i >> 6][i & 63];
}

// ---------------------------------------------------------------------------
// Host orchestration
// ---------------------------------------------------------------------------
extern "C" void kernel_launch(void* const* d_in, const int* in_sizes, int n_in,
                              void* d_out, int out_size, void* d_ws, size_t ws_size,
                              hipStream_t stream) {
  (void)in_sizes; (void)n_in; (void)out_size; (void)ws_size;
  static const int VERTS[5] = {5023, 1256, 314, 79, 20};

  const float* audio_hidden = (const float*)d_in[0];
  const float* actor        = (const float*)d_in[2];
  const int* sp_idx[4] = {(const int*)d_in[3], (const int*)d_in[4],
                          (const int*)d_in[5], (const int*)d_in[6]};
  const int   *d_rows[4], *d_cols[4];
  const float *d_vals[4];
  const int   *u_rows[4], *u_cols[4];
  const float *u_vals[4];
  for (int i = 0; i < 4; ++i) {
    d_rows[i] = (const int*)d_in[7 + 3 * i];
    d_cols[i] = (const int*)d_in[8 + 3 * i];
    d_vals[i] = (const float*)d_in[9 + 3 * i];
    u_rows[i] = (const int*)d_in[19 + 3 * i];
    u_cols[i] = (const int*)d_in[20 + 3 * i];
    u_vals[i] = (const float*)d_in[21 + 3 * i];
  }
  const float *enc_W[4], *enc_b[4];
  for (int i = 0; i < 4; ++i) {
    enc_W[i] = (const float*)d_in[31 + 2 * i];
    enc_b[i] = (const float*)d_in[32 + 2 * i];
  }
  const float* enc_lin_W = (const float*)d_in[39];
  const float* enc_lin_b = (const float*)d_in[40];
  const float* dec_lin_W = (const float*)d_in[41];
  const float* dec_lin_b = (const float*)d_in[42];
  const float *dec_W[5], *dec_b[5];
  for (int i = 0; i < 4; ++i) {
    dec_W[i] = (const float*)d_in[43 + 2 * i];
    dec_b[i] = (const float*)d_in[44 + 2 * i];
  }
  dec_W[4] = (const float*)d_in[51];
  dec_b[4] = (const float*)d_in[52];
  const float* aud_W = (const float*)d_in[53];
  const float* aud_b = (const float*)d_in[54];
  const float* l0_Wih  = (const float*)d_in[55];
  const float* l0_Whh  = (const float*)d_in[56];
  const float* l0_bih  = (const float*)d_in[57];
  const float* l0_bhh  = (const float*)d_in[58];
  const float* l12_Wih = (const float*)d_in[59];
  const float* l12_Whh = (const float*)d_in[60];
  const float* l12_bih = (const float*)d_in[61];
  const float* l12_bhh = (const float*)d_in[62];

  // Workspace layout (floats)
  float* ws = (float*)d_ws;
  const size_t BIG = (size_t)32 * 5023 * 64;  // 10,287,104 floats, max level buffer
  float* bufA      = ws;
  float* bufB      = ws + BIG;
  float* audio_emb = ws + 2 * BIG;          // 32*64
  float* zlat      = audio_emb + 32 * 64;   // 64
  float* h_in      = zlat + 64;             // 32*128
  float* latent    = h_in + 32 * 128;       // 32*64

  auto gemm = [&](const float* A, const float* W, const float* b, float* out,
                  int M, int N, int K) {
    const int pairs = (((M + 15) / 16 + 1) / 2) * (((N + 15) / 16 + 1) / 2);
    const int blocks = (pairs + WAVES_PER_BLOCK - 1) / WAVES_PER_BLOCK;
    wmma_gemm_nt<<<blocks, 32 * WAVES_PER_BLOCK, 0, stream>>>(A, W, b, out, M, N, K);
  };
  auto conv = [&](const float* x, const int* idx, const float* W, const float* b,
                  const float* res, float* out, int Bf, int Nv, int Cin, int Cout,
                  int elu) {
    const int M = Bf * Nv;
    if ((Cin & 7) == 0 && ((9 * Cin) & 31) == 0) {
      const int pairs = (((M + 15) / 16 + 1) / 2) * (((Cout + 15) / 16 + 1) / 2);
      const int blocks = (pairs + WAVES_PER_BLOCK - 1) / WAVES_PER_BLOCK;
      wmma_spiral_conv<<<blocks, 32 * WAVES_PER_BLOCK, 0, stream>>>(
          x, idx, W, b, res, out, Bf, Nv, Cin, Cout, elu);
    } else {
      const int tiles = ((M + 15) / 16) * ((Cout + 15) / 16);
      const int blocks = (tiles + WAVES_PER_BLOCK - 1) / WAVES_PER_BLOCK;
      wmma_spiral_conv_small<<<blocks, 32 * WAVES_PER_BLOCK, 0, stream>>>(
          x, idx, W, b, out, Bf, Nv, Cin, Cout, elu);
    }
  };
  auto pool = [&](const float* in, const int* rows, const int* cols,
                  const float* vals, float* out, int nnz, int C, int Bf,
                  int Nin, int Nout) {
    const int ztot = Bf * Nout * C;
    fill_zero<<<(ztot + 255) / 256, 256, 0, stream>>>(out, ztot);
    const int stot = nnz * C;
    pool_scatter<<<(stot + 255) / 256, 256, 0, stream>>>(in, rows, cols, vals,
                                                         out, nnz, C, Bf, Nin, Nout);
  };

  // --- Audio embedding: (32,1536) @ aud_W(64,1536)^T + aud_b ---
  gemm(audio_hidden, aud_W, aud_b, audio_emb, 32, 64, 1536);

  // --- Encoder (batch 1) ---
  static const int ENC_C[5] = {3, 32, 64, 128, 256};
  const float* ex = actor;
  for (int i = 0; i < 4; ++i) {
    conv(ex, sp_idx[i], enc_W[i], enc_b[i], nullptr, bufB,
         1, VERTS[i], ENC_C[i], ENC_C[i + 1], 1);
    pool(bufB, d_rows[i], d_cols[i], d_vals[i], bufA,
         3 * VERTS[i + 1], ENC_C[i + 1], 1, VERTS[i], VERTS[i + 1]);
    ex = bufA;
  }
  gemm(bufA, enc_lin_W, enc_lin_b, zlat, 1, 64, VERTS[4] * 256);

  // --- Concat + LSTM ---
  concat_kernel<<<(32 * 128 + 255) / 256, 256, 0, stream>>>(audio_emb, zlat, h_in);
  lstm_stack<<<1, 256, 0, stream>>>(h_in, l0_Wih, l0_Whh, l0_bih, l0_bhh,
                                    l12_Wih, l12_Whh, l12_bih, l12_bhh, latent);

  // --- Decoder (frame batch 32) ---
  gemm(latent, dec_lin_W, dec_lin_b, bufB, 32, VERTS[4] * 256, 64);  // (32,20,256)
  static const int DEC_CI[4] = {256, 256, 128, 64};
  static const int DEC_CO[4] = {256, 128, 64, 32};
  for (int i = 0; i < 4; ++i) {
    const int j = 3 - i;
    pool(bufB, u_rows[j], u_cols[j], u_vals[j], bufA,
         3 * VERTS[j], DEC_CI[i], 32, VERTS[j + 1], VERTS[j]);
    conv(bufA, sp_idx[j], dec_W[i], dec_b[i], nullptr, bufB,
         32, VERTS[j], DEC_CI[i], DEC_CO[i], 1);
  }
  // Final conv (no ELU) + actor residual -> d_out
  conv(bufB, sp_idx[0], dec_W[4], dec_b[4], actor, (float*)d_out,
       32, VERTS[0], 32, 3, 0);
}
